// RNN_83940840833741
// MI455X (gfx1250) — matrix operand order repaired and checked by reading, not compiled, not run on hardware
//
#include <hip/hip_runtime.h>
#include <hip/hip_bf16.h>
#include <math.h>

typedef __attribute__((ext_vector_type(16))) _Float16 v16h;
typedef __attribute__((ext_vector_type(8)))  _Float16 v8h;
typedef __attribute__((ext_vector_type(8)))  float    v8f;
typedef __attribute__((ext_vector_type(4)))  int      v4i;

// Address-space-qualified b128 pointers for the async global->LDS builtin.
typedef v4i __attribute__((address_space(1)))* gv4i_p;
typedef v4i __attribute__((address_space(3)))* lv4i_p;

#define HID   128
#define TLEN  2048
#define BATCH 4
#define VOCAB 32000
#define NTOK  (BATCH * TLEN)   // 8192

// ---------------------------------------------------------------------------
// CDNA5 async global->LDS helpers (guarded; fall back to VGPR-staged copy)
// ---------------------------------------------------------------------------
#if defined(__has_builtin)
#if __has_builtin(__builtin_amdgcn_global_load_async_to_lds_b128)
#define HAS_ASYNC_LDS 1
#endif
#endif

__device__ __forceinline__ void async_wait0() {
#if defined(__has_builtin) && __has_builtin(__builtin_amdgcn_s_wait_asynccnt)
    __builtin_amdgcn_s_wait_asynccnt(0);
#else
    asm volatile("s_wait_asynccnt 0x0" ::: "memory");
#endif
}

// Copy one contiguous 4KB tile (global f16 -> LDS) with 256 threads,
// one GLOBAL_LOAD_ASYNC_TO_LDS_B128 per lane (16B/lane, 512B/wave).
__device__ __forceinline__ void stage_tile_4k(const _Float16* __restrict__ gsrc,
                                              _Float16* __restrict__ lds,
                                              int tid) {
#ifdef HAS_ASYNC_LDS
    __builtin_amdgcn_global_load_async_to_lds_b128(
        (gv4i_p)(gsrc + (size_t)tid * 8),
        (lv4i_p)(lds + (size_t)tid * 8),
        0, 0);
    async_wait0();
#else
    *(v8h*)(lds + (size_t)tid * 8) = *(const v8h*)(gsrc + (size_t)tid * 8);
#endif
}

// ---------------------------------------------------------------------------
// f32 -> f16 conversion (fc_w once; W_ih0/W_ih1 once — avoids re-converting
// the B matrix inside every GEMM block)
// ---------------------------------------------------------------------------
__global__ __launch_bounds__(256)
void cvt_f32_to_f16_kernel(const float* __restrict__ src,
                           _Float16* __restrict__ dst, int n) {
    for (int i = blockIdx.x * blockDim.x + threadIdx.x; i < n;
         i += gridDim.x * blockDim.x) {
        dst[i] = (_Float16)src[i];
    }
}

// ---------------------------------------------------------------------------
// u[tok,f] = A[tok,:] @ W[f,:]^T + bias[f]      (f16 WMMA, f32 accumulate)
//   A source: either f16 activations Ah[tok,128], or emb[x[tok]][:] gather.
// One block = 16 tokens (M tile), 8 waves, wave w -> output cols [16w,16w+16).
// ---------------------------------------------------------------------------
__global__ __launch_bounds__(256)
void gemm_tok128_kernel(const _Float16* __restrict__ Ah,   // nullable
                        const int* __restrict__ xi,        // nullable
                        const float* __restrict__ emb,     // nullable
                        const _Float16* __restrict__ Bh,   // [128,128] f16
                        const float* __restrict__ bias,    // [128]
                        float* __restrict__ out) {         // [NTOK,128]
    __shared__ __attribute__((aligned(32))) _Float16 As[16 * HID];

    const int tid   = threadIdx.x;
    const int mtile = blockIdx.x;

    // Stage A tile (16 x 128 f16 = 4KB) into LDS.
    if (Ah) {
        stage_tile_4k(Ah + (size_t)mtile * 16 * HID, As, tid);
    } else {
        for (int e = tid; e < 16 * HID; e += 256) {
            int m = e >> 7, k = e & 127;
            int tok = mtile * 16 + m;
            As[e] = (_Float16)emb[(size_t)xi[tok] * HID + k];
        }
    }
    __syncthreads();

    const int wave = tid >> 5;
    const int lane = tid & 31;
    const int lh   = lane >> 4;       // half-wave select
    const int ll   = lane & 15;
    const int f    = wave * 16 + ll;  // output feature (B/C column)
    const int m    = ll;              // A row for this lane

    v8f c = {};
    #pragma unroll
    for (int ks = 0; ks < 4; ++ks) {
        const int kb = ks * 32;
        v16h a, b;
        // A fragment from LDS: two contiguous 8-half chunks.
        v8h alo = *(const v8h*)&As[m * HID + kb + lh * 16];
        v8h ahi = *(const v8h*)&As[m * HID + kb + lh * 16 + 8];
        #pragma unroll
        for (int p = 0; p < 8; ++p) { a[p] = alo[p]; a[8 + p] = ahi[p]; }
        // B fragment: 16 contiguous K halves from pre-converted f16 W row.
        const v8h* bp = (const v8h*)&Bh[(size_t)f * HID + kb + lh * 16];
        v8h blo = bp[0], bhi = bp[1];
        #pragma unroll
        for (int q = 0; q < 8; ++q) { b[q] = blo[q]; b[8 + q] = bhi[q]; }

        c = __builtin_amdgcn_wmma_f32_16x16x32_f16(
                false, a, false, b, (short)0, c, false, false);
    }

    const float bf = bias[f];
    #pragma unroll
    for (int r = 0; r < 8; ++r) {
        int row = lh * 8 + r;
        int tok = mtile * 16 + row;
        out[(size_t)tok * HID + f] = c[r] + bf;
    }
}

// ---------------------------------------------------------------------------
// Elman recurrence: h = tanh(u[:,t,:] + h @ W_hh^T + b_hh)
// The 4 batch chains are independent -> one workgroup per b (4 WGPs working
// in parallel on the serial critical path). 128 threads: thread f owns output
// feature f and keeps its whole W_hh row in VGPRs (1 wave/SIMD, registers are
// free). Per step: 32 broadcast LDS float4 loads of h + 128 FMAs + tanh,
// double-buffered h (single barrier/step), u[t+1] prefetched under the dot.
// ---------------------------------------------------------------------------
__global__ __launch_bounds__(128)
void rnn_rec_kernel(const float* __restrict__ u,        // [NTOK,128] (b*T+t)
                    const float* __restrict__ W_hh,     // [128,128]
                    const float* __restrict__ b_hh,     // [128]
                    const float* __restrict__ hidden_in,// [2,4,128]
                    int layer,
                    _Float16* __restrict__ yh,          // [NTOK,128] f16 out
                    float* __restrict__ h_out) {        // [2,4,128]
    __shared__ float hs[2][HID];

    const int f = threadIdx.x;   // feature 0..127
    const int b = blockIdx.x;    // batch chain 0..3

    // W_hh row f -> VGPRs (64KB read once per WG).
    float wreg[HID];
    #pragma unroll
    for (int k = 0; k < HID; ++k) wreg[k] = W_hh[(size_t)f * HID + k];

    hs[0][f] = hidden_in[layer * (BATCH * HID) + b * HID + f];
    __syncthreads();

    const float bh = b_hh[f];
    int cur = 0;
    float unext = u[(size_t)(b * TLEN) * HID + f];
    float hn = 0.0f;

    for (int t = 0; t < TLEN; ++t) {
        const float4* hv = (const float4*)hs[cur];
        float acc = 0.0f;
        #pragma unroll
        for (int q = 0; q < HID / 4; ++q) {
            float4 hq = hv[q];               // broadcast (same addr all lanes)
            acc = fmaf(hq.x, wreg[4 * q + 0], acc);
            acc = fmaf(hq.y, wreg[4 * q + 1], acc);
            acc = fmaf(hq.z, wreg[4 * q + 2], acc);
            acc = fmaf(hq.w, wreg[4 * q + 3], acc);
        }
        const size_t tok = (size_t)b * TLEN + t;
        hn = tanhf(unext + acc + bh);
        if (t + 1 < TLEN) unext = u[(tok + 1) * HID + f];  // prefetch next step

        hs[cur ^ 1][f] = hn;                 // other buffer: no WAR hazard
        yh[tok * HID + f] = (_Float16)hn;
        __syncthreads();                     // single barrier per step
        cur ^= 1;
    }
    h_out[layer * (BATCH * HID) + b * HID + f] = hn;
}

// ---------------------------------------------------------------------------
// logits[tok, v] = y1[tok,:] @ fc_w[v,:]^T + fc_b[v]   (f16 WMMA, f32 accum)
// grid = (512 M-tiles, 250 N-groups); block = 8 waves; wave w -> 16 columns.
// Logits streamed with non-temporal stores (write-once), keeping fc_w/y1h
// L2-resident while the 1.05 GB output goes straight to HBM.
// ---------------------------------------------------------------------------
__global__ __launch_bounds__(256)
void fc_gemm_kernel(const _Float16* __restrict__ Ah,   // [NTOK,128] f16
                    const _Float16* __restrict__ Bh,   // [VOCAB,128] f16
                    const float* __restrict__ fcb,     // [VOCAB]
                    float* __restrict__ logits) {      // [NTOK,VOCAB]
    __shared__ __attribute__((aligned(32))) _Float16 As[16 * HID];

    const int tid   = threadIdx.x;
    const int mtile = blockIdx.x;

    // Async-stage the contiguous 4KB A tile into LDS.
    stage_tile_4k(Ah + (size_t)mtile * 16 * HID, As, tid);
    __syncthreads();

    const int wave = tid >> 5;
    const int lane = tid & 31;
    const int lh   = lane >> 4;
    const int ll   = lane & 15;
    const int f    = (blockIdx.y * 8 + wave) * 16 + ll;  // vocab column
    const int m    = ll;

    v8f c = {};
    #pragma unroll
    for (int ks = 0; ks < 4; ++ks) {
        const int kb = ks * 32;
        v16h a, b;
        v8h alo = *(const v8h*)&As[m * HID + kb + lh * 16];
        v8h ahi = *(const v8h*)&As[m * HID + kb + lh * 16 + 8];
        #pragma unroll
        for (int p = 0; p < 8; ++p) { a[p] = alo[p]; a[8 + p] = ahi[p]; }
        // 16 contiguous K halves from fc_w row f (32B aligned).
        const v8h* bp = (const v8h*)&Bh[(size_t)f * HID + kb + lh * 16];
        v8h blo = bp[0], bhi = bp[1];
        #pragma unroll
        for (int q = 0; q < 8; ++q) { b[q] = blo[q]; b[8 + q] = bhi[q]; }

        c = __builtin_amdgcn_wmma_f32_16x16x32_f16(
                false, a, false, b, (short)0, c, false, false);
    }

    const float bf = fcb[f];
    #pragma unroll
    for (int r = 0; r < 8; ++r) {
        int row = lh * 8 + r;
        size_t tok = (size_t)(mtile * 16 + row);
        __builtin_nontemporal_store(c[r] + bf, &logits[tok * VOCAB + f]);
    }
}

// ---------------------------------------------------------------------------
extern "C" void kernel_launch(void* const* d_in, const int* in_sizes, int n_in,
                              void* d_out, int out_size, void* d_ws, size_t ws_size,
                              hipStream_t stream) {
    const int*   x      = (const int*)  d_in[0];
    const float* hidden = (const float*)d_in[1];
    const float* emb    = (const float*)d_in[2];
    const float* W_ih0  = (const float*)d_in[3];
    const float* W_hh0  = (const float*)d_in[4];
    const float* b_ih0  = (const float*)d_in[5];
    const float* b_hh0  = (const float*)d_in[6];
    const float* W_ih1  = (const float*)d_in[7];
    const float* W_hh1  = (const float*)d_in[8];
    const float* b_ih1  = (const float*)d_in[9];
    const float* b_hh1  = (const float*)d_in[10];
    const float* fc_w   = (const float*)d_in[11];
    const float* fc_b   = (const float*)d_in[12];

    float* logits = (float*)d_out;
    float* h_out  = (float*)d_out + (size_t)NTOK * VOCAB;  // [2,4,128]

    char* ws = (char*)d_ws;
    float*    u0    = (float*)   (ws + 0);         //  4 MB
    float*    u1    = (float*)   (ws + 4194304);   //  4 MB
    _Float16* y0h   = (_Float16*)(ws + 8388608);   //  2 MB
    _Float16* y1h   = (_Float16*)(ws + 10485760);  //  2 MB
    _Float16* fcwh  = (_Float16*)(ws + 12582912);  //  8 MB
    _Float16* wih0h = (_Float16*)(ws + 20774912);  // 32 KB
    _Float16* wih1h = (_Float16*)(ws + 20807680);  // 32 KB

    // 1) one-time f32 -> f16 weight conversions (all L2-resident afterwards)
    cvt_f32_to_f16_kernel<<<2000, 256, 0, stream>>>(fc_w,  fcwh,  VOCAB * HID);
    cvt_f32_to_f16_kernel<<<64,   256, 0, stream>>>(W_ih0, wih0h, HID * HID);
    cvt_f32_to_f16_kernel<<<64,   256, 0, stream>>>(W_ih1, wih1h, HID * HID);

    // 2) u0 = emb[x] @ W_ih0^T + b_ih0   (fused gather + WMMA GEMM)
    gemm_tok128_kernel<<<NTOK / 16, 256, 0, stream>>>(
        nullptr, x, emb, wih0h, b_ih0, u0);

    // 3) layer-0 recurrence -> y0h, h_out[0]   (4 WGs, one per batch chain)
    rnn_rec_kernel<<<BATCH, HID, 0, stream>>>(u0, W_hh0, b_hh0, hidden, 0, y0h, h_out);

    // 4) u1 = y0 @ W_ih1^T + b_ih1
    gemm_tok128_kernel<<<NTOK / 16, 256, 0, stream>>>(
        y0h, nullptr, nullptr, wih1h, b_ih1, u1);

    // 5) layer-1 recurrence -> y1h, h_out[1]
    rnn_rec_kernel<<<BATCH, HID, 0, stream>>>(u1, W_hh1, b_hh1, hidden, 1, y1h, h_out);

    // 6) logits = y1 @ fc_w^T + fc_b   (store-bandwidth-bound WMMA GEMM)
    fc_gemm_kernel<<<dim3(NTOK / 16, VOCAB / 128), 256, 0, stream>>>(
        y1h, fcwh, fc_b, logits);
}
